// TypeSpecificGATLayer_43087111914088
// MI455X (gfx1250) — matrix operand order repaired and checked by reading, not compiled, not run on hardware
//
#include <hip/hip_runtime.h>
#include <hip/hip_bf16.h>

#define IN_DIM 256
#define OUT_DIM 256
#define SLOPE 0.2f

typedef __attribute__((ext_vector_type(16))) _Float16 v16h;
typedef __attribute__((ext_vector_type(8)))  float    v8f;
typedef __attribute__((ext_vector_type(4)))  float    f4;

union FragAB { v16h v; f4 f[2]; };

// ---------------------------------------------------------------- converts
__global__ __launch_bounds__(256) void cvt_h_kernel(const float* __restrict__ H,
                                                    _Float16* __restrict__ Hh, int total) {
    int i = blockIdx.x * 256 + threadIdx.x;
    if (i < total) Hh[i] = (_Float16)H[i];
}

// WT[t][n*256+k] = W[t][k*256+n]  (f16, transposed so B-fragment loads are contiguous)
__global__ __launch_bounds__(256) void cvt_w_kernel(const float* __restrict__ W0,
                                                    const float* __restrict__ W1,
                                                    const float* __restrict__ W2,
                                                    _Float16* __restrict__ WT) {
    int idx = blockIdx.x * 256 + threadIdx.x;          // 0 .. 3*65536
    int t   = idx >> 16;
    int rem = idx & 65535;
    int n   = rem >> 8;
    int k   = rem & 255;
    const float* W = (t == 0) ? W0 : ((t == 1) ? W1 : W2);
    WT[idx] = (_Float16)W[k * 256 + n];
}

__global__ __launch_bounds__(256) void init_out_kernel(float* __restrict__ out,
                                                       const float* __restrict__ bias, int total) {
    int i = blockIdx.x * 256 + threadIdx.x;
    if (i < total) out[i] = bias[i & 255];
}

__global__ __launch_bounds__(256) void init_ms_kernel(float* __restrict__ emax,
                                                      float* __restrict__ esum, int Nn) {
    int i = blockIdx.x * 256 + threadIdx.x;
    if (i < Nn) { emax[i] = __int_as_float(0xff800000); esum[i] = 0.0f; }
}

// ---------------------------------------------------------------- WMMA GEMM
// Wh[N,256] = Hh[N,256] @ W[256,256]   (W pre-transposed as WT[n][k], f16)
// block: 256 threads = 8 waves; block tile 32 rows x 256 cols; wave tile 16x64.
__global__ __launch_bounds__(256) void gat_gemm_wmma_kernel(const _Float16* __restrict__ Hh,
                                                            const _Float16* __restrict__ WT,
                                                            float* __restrict__ Wh, int Nn) {
    const int tid  = threadIdx.x;
    const int lane = tid & 31;
    const int wave = tid >> 5;
    const int half = lane >> 4;   // 0 or 1
    const int l16  = lane & 15;

    const int rowBase = blockIdx.x * 32 + (wave >> 2) * 16;
    const int colBase = (wave & 3) * 64;

    v8f acc0 = {}, acc1 = {}, acc2 = {}, acc3 = {};

    int arow = rowBase + l16;
    if (arow >= Nn) arow = Nn - 1;               // clamp (N is a multiple of 32 anyway)
    const _Float16* Abase = Hh + (size_t)arow * 256;
    const _Float16* Bbase = WT + (size_t)(colBase + l16) * 256;

#pragma unroll
    for (int kb = 0; kb < 8; ++kb) {
        const int k0 = kb * 32;

        // A fragment: lane holds row M=l16; VGPR0..7 = K[half*16 .. +15]
        FragAB A;
        const _Float16* ap = Abase + k0 + half * 16;
        A.f[0] = *(const f4*)(ap);
        A.f[1] = *(const f4*)(ap + 8);

        // B fragments: lane holds col N=l16 of each tile; lanes 0-15 K[0..15], 16-31 K[16..31]
        const _Float16* bp = Bbase + k0 + half * 16;
        FragAB B0, B1, B2, B3;
        B0.f[0] = *(const f4*)(bp);          B0.f[1] = *(const f4*)(bp + 8);
        B1.f[0] = *(const f4*)(bp + 4096);   B1.f[1] = *(const f4*)(bp + 4096 + 8);
        B2.f[0] = *(const f4*)(bp + 8192);   B2.f[1] = *(const f4*)(bp + 8192 + 8);
        B3.f[0] = *(const f4*)(bp + 12288);  B3.f[1] = *(const f4*)(bp + 12288 + 8);

        acc0 = __builtin_amdgcn_wmma_f32_16x16x32_f16(false, A.v, false, B0.v, (short)0, acc0, false, false);
        acc1 = __builtin_amdgcn_wmma_f32_16x16x32_f16(false, A.v, false, B1.v, (short)0, acc1, false, false);
        acc2 = __builtin_amdgcn_wmma_f32_16x16x32_f16(false, A.v, false, B2.v, (short)0, acc2, false, false);
        acc3 = __builtin_amdgcn_wmma_f32_16x16x32_f16(false, A.v, false, B3.v, (short)0, acc3, false, false);
    }

    // C/D layout: VGPR r -> row = r + half*8, col = l16 (within 16x16 tile)
    const int crow0 = rowBase + half * 8;
#pragma unroll
    for (int r = 0; r < 8; ++r) {
        const int row = crow0 + r;
        if (row < Nn) {
            float* o = Wh + (size_t)row * 256 + colBase + l16;
            o[0]  = acc0[r];
            o[16] = acc1[r];
            o[32] = acc2[r];
            o[48] = acc3[r];
        }
    }
}

// ---------------------------------------------------------------- per-node scores
// one wave per node: s_dst = Wh[node]·a[0:256], s_src = Wh[node]·a[256:512]
__global__ __launch_bounds__(256) void scores_kernel(const float* __restrict__ Wh,
                                                     const float* __restrict__ a,
                                                     float* __restrict__ sdst,
                                                     float* __restrict__ ssrc, int Nn) {
    const int lane = threadIdx.x & 31;
    const int node = blockIdx.x * 8 + (threadIdx.x >> 5);
    if (node >= Nn) return;

    const f4* w  = (const f4*)(Wh + (size_t)node * 256) + lane * 2;
    const f4* ad = (const f4*)(a)       + lane * 2;
    const f4* as = (const f4*)(a + 256) + lane * 2;
    f4 w0 = w[0], w1 = w[1];
    f4 d0 = ad[0], d1 = ad[1], s0 = as[0], s1 = as[1];

    float sd = w0.x*d0.x + w0.y*d0.y + w0.z*d0.z + w0.w*d0.w
             + w1.x*d1.x + w1.y*d1.y + w1.z*d1.z + w1.w*d1.w;
    float ss = w0.x*s0.x + w0.y*s0.y + w0.z*s0.z + w0.w*s0.w
             + w1.x*s1.x + w1.y*s1.y + w1.z*s1.z + w1.w*s1.w;

#pragma unroll
    for (int off = 16; off; off >>= 1) {
        sd += __shfl_xor(sd, off, 32);
        ss += __shfl_xor(ss, off, 32);
    }
    if (lane == 0) { sdst[node] = sd; ssrc[node] = ss; }
}

// ---------------------------------------------------------------- edge softmax passes
__device__ __forceinline__ float edge_logit(const float* sdst, const float* ssrc, int r, int c) {
    float s = sdst[r] + ssrc[c];
    return s > 0.0f ? s : SLOPE * s;
}

// sign-aware float atomic max via integer punning (emax initialized to -inf bits)
__device__ __forceinline__ void atomicMaxF(float* addr, float v) {
    if (v >= 0.0f) atomicMax((int*)addr, __float_as_int(v));
    else           atomicMin((unsigned int*)addr, __float_as_uint(v));
}

__global__ __launch_bounds__(256) void edge_max_kernel(const int* __restrict__ row,
                                                       const int* __restrict__ col,
                                                       const float* __restrict__ sdst,
                                                       const float* __restrict__ ssrc,
                                                       float* __restrict__ emax, int Ee) {
    int e = blockIdx.x * 256 + threadIdx.x;
    if (e >= Ee) return;
    int r = row[e], c = col[e];
    atomicMaxF(&emax[r], edge_logit(sdst, ssrc, r, c));
}

__global__ __launch_bounds__(256) void edge_sum_kernel(const int* __restrict__ row,
                                                       const int* __restrict__ col,
                                                       const float* __restrict__ sdst,
                                                       const float* __restrict__ ssrc,
                                                       const float* __restrict__ emax,
                                                       float* __restrict__ esum, int Ee) {
    int e = blockIdx.x * 256 + threadIdx.x;
    if (e >= Ee) return;
    int r = row[e], c = col[e];
    atomicAdd(&esum[r], __expf(edge_logit(sdst, ssrc, r, c) - emax[r]));
}

// one wave per edge: out[row] += alpha * Wh[col]  (Wh + out are L2-resident: ~100 MB each)
__global__ __launch_bounds__(256) void edge_agg_kernel(const int* __restrict__ row,
                                                       const int* __restrict__ col,
                                                       const float* __restrict__ sdst,
                                                       const float* __restrict__ ssrc,
                                                       const float* __restrict__ emax,
                                                       const float* __restrict__ esum,
                                                       const float* __restrict__ Wh,
                                                       float* __restrict__ out, int Ee) {
    const int lane = threadIdx.x & 31;
    const int e = blockIdx.x * 8 + (threadIdx.x >> 5);
    if (e >= Ee) return;
    const int r = row[e], c = col[e];
    const float w = __expf(edge_logit(sdst, ssrc, r, c) - emax[r]);
    const float alpha = w / fmaxf(esum[r], 1e-12f);

    const f4* src = (const f4*)(Wh + (size_t)c * 256) + lane * 2;
    f4 v0 = src[0], v1 = src[1];
    float* dst = out + (size_t)r * 256 + lane * 8;
    atomicAdd(dst + 0, alpha * v0.x);
    atomicAdd(dst + 1, alpha * v0.y);
    atomicAdd(dst + 2, alpha * v0.z);
    atomicAdd(dst + 3, alpha * v0.w);
    atomicAdd(dst + 4, alpha * v1.x);
    atomicAdd(dst + 5, alpha * v1.y);
    atomicAdd(dst + 6, alpha * v1.z);
    atomicAdd(dst + 7, alpha * v1.w);
}

// ---------------------------------------------------------------- host launch
extern "C" void kernel_launch(void* const* d_in, const int* in_sizes, int n_in,
                              void* d_out, int out_size, void* d_ws, size_t ws_size,
                              hipStream_t stream) {
    const float* H    = (const float*)d_in[0];
    const float* W[3] = {(const float*)d_in[1], (const float*)d_in[2], (const float*)d_in[3]};
    const float* a[3] = {(const float*)d_in[4], (const float*)d_in[5], (const float*)d_in[6]};
    const float* bias = (const float*)d_in[7];
    const int* row[3] = {(const int*)d_in[8], (const int*)d_in[10], (const int*)d_in[12]};
    const int* col[3] = {(const int*)d_in[9], (const int*)d_in[11], (const int*)d_in[13]};
    float* out = (float*)d_out;

    const int Ntot = in_sizes[0];          // N * 256
    const int N    = Ntot / 256;
    const int E    = in_sizes[8];

    auto alignup = [](size_t x) { return (x + 255) & ~(size_t)255; };
    char* ws = (char*)d_ws;
    size_t off = 0;
    _Float16* Hh = (_Float16*)(ws + off); off = alignup(off + (size_t)Ntot * 2);
    _Float16* WT = (_Float16*)(ws + off); off = alignup(off + (size_t)3 * 256 * 256 * 2);
    float* Wh   = (float*)(ws + off); off = alignup(off + (size_t)Ntot * 4);
    float* sdst = (float*)(ws + off); off = alignup(off + (size_t)N * 4);
    float* ssrc = (float*)(ws + off); off = alignup(off + (size_t)N * 4);
    float* emax = (float*)(ws + off); off = alignup(off + (size_t)N * 4);
    float* esum = (float*)(ws + off); off = alignup(off + (size_t)N * 4);

    const int thr = 256;
    cvt_h_kernel<<<(Ntot + thr - 1) / thr, thr, 0, stream>>>(H, Hh, Ntot);
    cvt_w_kernel<<<(3 * 65536) / thr, thr, 0, stream>>>(W[0], W[1], W[2], WT);
    init_out_kernel<<<(Ntot + thr - 1) / thr, thr, 0, stream>>>(out, bias, Ntot);

    const int gemmBlocks  = (N + 31) / 32;
    const int nodeWaveBlk = (N + 7) / 8;
    const int edgeThrBlk  = (E + thr - 1) / thr;
    const int edgeWaveBlk = (E + 7) / 8;

    for (int t = 0; t < 3; ++t) {
        gat_gemm_wmma_kernel<<<gemmBlocks, thr, 0, stream>>>(Hh, WT + (size_t)t * 65536, Wh, N);
        scores_kernel<<<nodeWaveBlk, thr, 0, stream>>>(Wh, a[t], sdst, ssrc, N);
        init_ms_kernel<<<(N + thr - 1) / thr, thr, 0, stream>>>(emax, esum, N);
        edge_max_kernel<<<edgeThrBlk, thr, 0, stream>>>(row[t], col[t], sdst, ssrc, emax, E);
        edge_sum_kernel<<<edgeThrBlk, thr, 0, stream>>>(row[t], col[t], sdst, ssrc, emax, esum, E);
        edge_agg_kernel<<<edgeWaveBlk, thr, 0, stream>>>(row[t], col[t], sdst, ssrc, emax, esum,
                                                         Wh, out, E);
    }
}
